// TransformerBlock_63995012710905
// MI455X (gfx1250) — matrix operand order repaired and checked by reading, not compiled, not run on hardware
//
#include <hip/hip_runtime.h>
#include <hip/hip_bf16.h>
#include <stdint.h>

// ---------------------------------------------------------------------------
// Transformer block on MI455X (gfx1250), bf16 WMMA pipeline + TDM staging.
//   out = mid + MLP(mid),  mid = 2 * W_O · Attn(x),  Attn per reference's
//   (buggy-but-faithful) mask:  p>q -> 0, p==q -> score, p<q -> score-1e10,
//   softmax over full row with 1/sqrt(DH) scale.
// GEMMs: C[M,N] = A[M,K] (bf16 row-major) x W[N,K]^T (bf16 row-major),
// f32 accumulate via v_wmma_f32_16x16x32_bf16; tiles staged into LDS by the
// Tensor Data Mover (tensor_load_to_lds + s_wait_tensorcnt).
// ---------------------------------------------------------------------------

typedef __bf16 bf16;
typedef __attribute__((ext_vector_type(8)))  __bf16 v8bf;
typedef __attribute__((ext_vector_type(16))) __bf16 v16bf;
typedef __attribute__((ext_vector_type(8)))  float  v8f;
typedef __attribute__((ext_vector_type(4)))  unsigned int u32x4;
typedef __attribute__((ext_vector_type(4)))  int i32x4;
typedef __attribute__((ext_vector_type(8)))  int i32x8;

#define B_   2
#define S_   2048
#define D_   1024
#define H_   16
#define DH_  64
#define HDH_ (H_ * DH_)   // 1024
#define DM_  4096
#define LSTR 72           // LDS row stride (bf16): 144B rows, 16B aligned

// --- TDM availability / arity dispatch -------------------------------------
#if defined(__has_builtin)
#if __has_builtin(__builtin_amdgcn_tensor_load_to_lds)
#define HAVE_TDM 1
#else
#define HAVE_TDM 0
#endif
#else
#define HAVE_TDM 0
#endif

#if __has_include(<hip/amd_detail/amd_gfx1250_TDM.h>)
#define TDM_SIX_ARGS 1   // therock-10.0 headers -> 6-arg builtin
#else
#define TDM_SIX_ARGS 0   // ROCm 7.2 -> 5-arg builtin
#endif

static __device__ inline v16bf join16(v8bf lo, v8bf hi) {
  v16bf r;
#pragma unroll
  for (int i = 0; i < 8; ++i) { r[i] = lo[i]; r[i + 8] = hi[i]; }
  return r;
}

// A fragment 16x32 bf16 (rows m=lane&15, cols k0..k0+31), row-major src.
// Layout: lane half hi holds K = hi*16 + e (16 contiguous bf16 per lane).
static __device__ inline v16bf load_a_frag(const bf16* A, int lda, int k0) {
  const int lane = threadIdx.x & 31;
  const int m = lane & 15, hi = lane >> 4;
  const bf16* row = A + (size_t)m * lda + k0;
  v8bf lo = *(const v8bf*)(row + hi * 16);
  v8bf h8 = *(const v8bf*)(row + hi * 16 + 8);
  return join16(lo, h8);
}

// B fragment 32x16: B[k][n] = W[n0+n][k0+k], W row-major stride ldw.
// Layout (same as A): lane n, half hi holds K = hi*16 + e (16 contiguous bf16 per lane).
static __device__ inline v16bf load_b_frag(const bf16* W, int ldw, int n0, int k0) {
  const int lane = threadIdx.x & 31;
  const int n = lane & 15, hi = lane >> 4;
  const bf16* row = W + (size_t)(n0 + n) * ldw + k0 + hi * 16;
  v8bf lo = *(const v8bf*)(row);
  v8bf h8 = *(const v8bf*)(row + 8);
  return join16(lo, h8);
}

static __device__ inline v8f wmma_bf16(v16bf a, v16bf b, v8f c) {
  return __builtin_amdgcn_wmma_f32_16x16x32_bf16(
      false, a, false, b, (short)0, c, false, false);
}

// --- TDM: DMA a 2D tile (tile_w x tile_h bf16, row stride = row_stride elems)
// from global into LDS at byte offset lds_off, padded +8 bf16 per 64-elem row
// (pad_interval=32 dwords, pad_amount=4 dwords) -> LDS row stride LSTR=72.
#if HAVE_TDM
static __device__ inline void tdm_load_2d(const bf16* gptr, unsigned lds_off,
                                          unsigned tile_w, unsigned tile_h,
                                          unsigned long long row_stride) {
  const unsigned long long ga = (unsigned long long)(uintptr_t)gptr;
  u32x4 g0;
  g0[0] = 1u;                                         // count=1 (valid user D#)
  g0[1] = lds_off;                                    // lds_addr [63:32]
  g0[2] = (unsigned)(ga & 0xffffffffu);               // global_addr [95:64]
  g0[3] = (unsigned)((ga >> 32) & 0x01ffffffu)        // global_addr [120:96]
        | (2u << 30);                                 // type=2 ("image")
  i32x8 g1;
  g1[0] = (int)((1u << 16)      // data_size = 1 -> 2 bytes
              | (1u << 20)      // pad_enable
              | (4u << 22)      // pad_interval: 32 dwords (one 64-elem row)
              | (3u << 25));    // pad_amount: 4 dwords (8 bf16)
  g1[1] = (int)((tile_w & 0xffffu) << 16);            // tensor_dim0 [79:48]
  g1[2] = (int)((tile_w >> 16) | ((tile_h & 0xffffu) << 16)); // dim0 hi, dim1 lo
  g1[3] = (int)((tile_h >> 16) | (tile_w << 16));     // dim1 hi, tile_dim0
  g1[4] = (int)(tile_h & 0xffffu);                    // tile_dim1 (tile_dim2=0)
  g1[5] = (int)(row_stride & 0xffffffffu);            // tensor_dim0_stride lo
  g1[6] = (int)((row_stride >> 32) & 0xffffu);        // stride hi (dim1_stride=0)
  g1[7] = 0;
  i32x4 z4 = {};
#if TDM_SIX_ARGS
  i32x8 z8 = {};
  __builtin_amdgcn_tensor_load_to_lds(g0, g1, z4, z4, z8, 0);
#else
  __builtin_amdgcn_tensor_load_to_lds(g0, g1, z4, z4, 0);
#endif
}
#endif

// ---------------------------------------------------------------------------
__global__ void cvt_f32_bf16(const float* __restrict__ src, bf16* __restrict__ dst, int n) {
  int i = blockIdx.x * blockDim.x + threadIdx.x;
  if (i < n) dst[i] = (bf16)src[i];
}

// ---------------------------------------------------------------------------
// bf16 WMMA GEMM, TDM-fed. Block = 128 threads (4 waves). Block tile 128x64;
// each wave owns a 32-row strip x 64 cols (8 C fragments). grid=(N/64, M/128).
__global__ void gemm_bf16_wmma(const bf16* __restrict__ A, int lda,
                               const bf16* __restrict__ W, int ldw,
                               int N, int K,
                               const float* __restrict__ bias,
                               float scale, int relu,
                               float* __restrict__ outF, int addF,
                               bf16* __restrict__ outB) {
  __shared__ alignas(16) bf16 Atile[128 * LSTR];   // 18.0 KB
  __shared__ alignas(16) bf16 Wtile[64 * LSTR];    //  9.0 KB
  const int lane = threadIdx.x & 31;
  const int wid  = threadIdx.x >> 5;               // 0..3
  const int m0 = blockIdx.y * 128;
  const int n0 = blockIdx.x * 64;
  v8f c[2][4] = {};
#if HAVE_TDM
  const unsigned aoff = (unsigned)(uintptr_t)(void*)Atile;
  const unsigned woff = (unsigned)(uintptr_t)(void*)Wtile;
#endif

  for (int k0 = 0; k0 < K; k0 += 64) {
#if HAVE_TDM
    if (threadIdx.x < 32) {                        // wave 0 drives the TDM
      tdm_load_2d(A + (size_t)m0 * lda + k0, aoff, 64u, 128u,
                  (unsigned long long)lda);
      tdm_load_2d(W + (size_t)n0 * ldw + k0, woff, 64u, 64u,
                  (unsigned long long)ldw);
      __builtin_amdgcn_s_wait_tensorcnt(0);
    }
#else
    for (int idx = threadIdx.x; idx < 128 * 64; idx += blockDim.x) {
      const int rr = idx >> 6, cc = idx & 63;
      Atile[rr * LSTR + cc] = A[(size_t)(m0 + rr) * lda + k0 + cc];
    }
    for (int idx = threadIdx.x; idx < 64 * 64; idx += blockDim.x) {
      const int rr = idx >> 6, cc = idx & 63;
      Wtile[rr * LSTR + cc] = W[(size_t)(n0 + rr) * ldw + k0 + cc];
    }
#endif
    __syncthreads();
#pragma unroll
    for (int kk = 0; kk < 64; kk += 32) {
      v16bf a0 = load_a_frag(Atile + (size_t)(wid * 32) * LSTR, LSTR, kk);
      v16bf a1 = load_a_frag(Atile + (size_t)(wid * 32 + 16) * LSTR, LSTR, kk);
#pragma unroll
      for (int ns = 0; ns < 4; ++ns) {
        v16bf b = load_b_frag(Wtile, LSTR, ns * 16, kk);
        c[0][ns] = wmma_bf16(a0, b, c[0][ns]);
        c[1][ns] = wmma_bf16(a1, b, c[1][ns]);
      }
    }
    __syncthreads();
  }

  const int nl = lane & 15, hi = lane >> 4;
#pragma unroll
  for (int sub = 0; sub < 2; ++sub)
#pragma unroll
    for (int ns = 0; ns < 4; ++ns) {
      const int col = n0 + ns * 16 + nl;
      const float bv = bias ? bias[col] : 0.0f;
#pragma unroll
      for (int r = 0; r < 8; ++r) {
        const int row = m0 + wid * 32 + sub * 16 + hi * 8 + r;
        float v = c[sub][ns][r] * scale + bv;
        if (relu) v = fmaxf(v, 0.0f);
        const size_t idx = (size_t)row * N + col;
        if (outF) { if (addF) outF[idx] += v; else outF[idx] = v; }
        if (outB) outB[idx] = (bf16)v;
      }
    }
}

// ---------------------------------------------------------------------------
// Fused attention: grid = (S/64, H, B), block = 128 (4 waves).
// Each wave: one 16-row q strip. Online softmax over ALL p tiles (reference
// mask: p>q -> 0, p==q -> s, p<q -> s-1e10, then *0.125).
__global__ void attn_kernel(const bf16* __restrict__ Qb,
                            const bf16* __restrict__ Kb,
                            const bf16* __restrict__ Vb,
                            bf16* __restrict__ Zb) {
  __shared__ alignas(16) bf16 Vt[64 * LSTR];        // V^T tile: [dh][p]
  __shared__ alignas(16) bf16 Pst[4][16 * LSTR];    // per-wave P strip [q][p]

  const int lane = threadIdx.x & 31;
  const int wid  = threadIdx.x >> 5;
  const int head = blockIdx.y;
  const int b    = blockIdx.z;
  const int q0   = blockIdx.x * 64 + wid * 16;
  const size_t base = ((size_t)b * S_) * HDH_ + head * DH_;

  const bf16* Aq = Qb + base + (size_t)q0 * HDH_;
  v16bf qf[2];
  qf[0] = load_a_frag(Aq, HDH_, 0);
  qf[1] = load_a_frag(Aq, HDH_, 32);

  const int nl = lane & 15, hi = lane >> 4;
  float Mrow[8], Lrow[8];
  v8f O[4] = {};
#pragma unroll
  for (int r = 0; r < 8; ++r) { Mrow[r] = -1e30f; Lrow[r] = 0.0f; }
  bf16* Pw = &Pst[wid][0];

  for (int p0 = 0; p0 < S_; p0 += 64) {
    for (int idx = threadIdx.x; idx < 64 * 64; idx += blockDim.x) {
      const int p = idx >> 6, dh = idx & 63;
      Vt[dh * LSTR + p] = Vb[base + (size_t)(p0 + p) * HDH_ + dh];
    }
    __syncthreads();

    const bf16* Kp = Kb + base + (size_t)p0 * HDH_;
    float sc[4][8];
#pragma unroll
    for (int ns = 0; ns < 4; ++ns) {
      v8f c = {};
      c = wmma_bf16(qf[0], load_b_frag(Kp, HDH_, ns * 16, 0), c);
      c = wmma_bf16(qf[1], load_b_frag(Kp, HDH_, ns * 16, 32), c);
      const int p = p0 + ns * 16 + nl;
#pragma unroll
      for (int r = 0; r < 8; ++r) {
        const int q = q0 + hi * 8 + r;
        const float v = c[r];
        const float m = (p > q) ? 0.0f : ((p == q) ? v : (v - 1e10f));
        sc[ns][r] = m * 0.125f;                      // 1/sqrt(64)
      }
    }

    float Mnew[8], alpha[8];
#pragma unroll
    for (int r = 0; r < 8; ++r) {
      float mx = fmaxf(fmaxf(sc[0][r], sc[1][r]), fmaxf(sc[2][r], sc[3][r]));
#pragma unroll
      for (int s = 1; s < 16; s <<= 1) mx = fmaxf(mx, __shfl_xor(mx, s));
      Mnew[r]  = fmaxf(Mrow[r], mx);
      alpha[r] = __expf(Mrow[r] - Mnew[r]);
      Mrow[r]  = Mnew[r];
    }
#pragma unroll
    for (int ns = 0; ns < 4; ++ns)
#pragma unroll
      for (int r = 0; r < 8; ++r) {
        const float e = __expf(sc[ns][r] - Mnew[r]);
        sc[ns][r] = e;
        Pw[(hi * 8 + r) * LSTR + ns * 16 + nl] = (bf16)e;
      }
#pragma unroll
    for (int r = 0; r < 8; ++r) {
      float s = sc[0][r] + sc[1][r] + sc[2][r] + sc[3][r];
#pragma unroll
      for (int m = 1; m < 16; m <<= 1) s += __shfl_xor(s, m);
      Lrow[r] = Lrow[r] * alpha[r] + s;
    }
#pragma unroll
    for (int ns = 0; ns < 4; ++ns)
#pragma unroll
      for (int r = 0; r < 8; ++r) O[ns][r] *= alpha[r];

#pragma unroll
    for (int kc = 0; kc < 2; ++kc) {
      const bf16* Prow = Pw + (size_t)nl * LSTR + kc * 32;
      v8bf plo = *(const v8bf*)(Prow + hi * 16);
      v8bf phi = *(const v8bf*)(Prow + hi * 16 + 8);
      v16bf pa = join16(plo, phi);
#pragma unroll
      for (int ns = 0; ns < 4; ++ns) {
        const bf16* Vrow = Vt + (size_t)(ns * 16 + nl) * LSTR + kc * 32 + hi * 16;
        v8bf bl = *(const v8bf*)Vrow;
        v8bf bh = *(const v8bf*)(Vrow + 8);
        O[ns] = wmma_bf16(pa, join16(bl, bh), O[ns]);
      }
    }
    __syncthreads();
  }

#pragma unroll
  for (int ns = 0; ns < 4; ++ns)
#pragma unroll
    for (int r = 0; r < 8; ++r) {
      const float z = O[ns][r] / Lrow[r];
      Zb[base + (size_t)(q0 + hi * 8 + r) * HDH_ + ns * 16 + nl] = (bf16)z;
    }
}

// ---------------------------------------------------------------------------
extern "C" void kernel_launch(void* const* d_in, const int* in_sizes, int n_in,
                              void* d_out, int out_size, void* d_ws, size_t ws_size,
                              hipStream_t stream) {
  const float* x     = (const float*)d_in[0];
  const float* W_Q   = (const float*)d_in[1];
  const float* W_K   = (const float*)d_in[2];
  const float* W_V   = (const float*)d_in[3];
  const float* W_O   = (const float*)d_in[4];
  const float* W_in  = (const float*)d_in[5];
  const float* b_in  = (const float*)d_in[6];
  const float* W_out = (const float*)d_in[7];
  const float* b_out = (const float*)d_in[8];
  float* out = (float*)d_out;

  const size_t nX  = (size_t)B_ * S_ * D_;    // 4M
  const size_t nW1 = (size_t)HDH_ * D_;       // 1M
  const size_t nWi = (size_t)DM_ * D_;        // 4M
  const size_t nH  = (size_t)B_ * S_ * DM_;   // 16M

  char* ws = (char*)d_ws;
  size_t off = 0;
  auto carve = [&](size_t elems) -> bf16* {
    bf16* p = (bf16*)(ws + off);
    off += (elems * sizeof(bf16) + 255) & ~(size_t)255;
    return p;
  };
  bf16* xb    = carve(nX);
  bf16* Wqb   = carve(nW1);
  bf16* Wkb   = carve(nW1);
  bf16* Wvb   = carve(nW1);
  bf16* Wob   = carve(nW1);
  bf16* Winb  = carve(nWi);
  bf16* Woutb = carve(nWi);
  bf16* Qb    = carve(nX);
  bf16* Kb    = carve(nX);
  bf16* Vb    = carve(nX);
  bf16* Zb    = carve(nX);
  bf16* midb  = carve(nX);
  bf16* hb    = carve(nH);
  (void)ws_size; (void)in_sizes; (void)n_in; (void)out_size;  // ~110 MB used

  auto cvt = [&](const float* s, bf16* d, size_t n) {
    cvt_f32_bf16<<<(unsigned)((n + 255) / 256), 256, 0, stream>>>(s, d, (int)n);
  };
  cvt(x, xb, nX);
  cvt(W_Q, Wqb, nW1);  cvt(W_K, Wkb, nW1);  cvt(W_V, Wvb, nW1);
  cvt(W_O, Wob, nW1);  cvt(W_in, Winb, nWi); cvt(W_out, Woutb, nWi);

  const int M = B_ * S_;  // 4096
  // Q/K/V projections: [4096,1024] x [1024,1024]^T -> bf16
  dim3 gP(HDH_ / 64, M / 128);
  gemm_bf16_wmma<<<gP, 128, 0, stream>>>(xb, D_, Wqb, D_, HDH_, D_,
                                         nullptr, 1.0f, 0, nullptr, 0, Qb);
  gemm_bf16_wmma<<<gP, 128, 0, stream>>>(xb, D_, Wkb, D_, HDH_, D_,
                                         nullptr, 1.0f, 0, nullptr, 0, Kb);
  gemm_bf16_wmma<<<gP, 128, 0, stream>>>(xb, D_, Wvb, D_, HDH_, D_,
                                         nullptr, 1.0f, 0, nullptr, 0, Vb);

  // Fused attention -> Zb (z_flat layout)
  attn_kernel<<<dim3(S_ / 64, H_, B_), 128, 0, stream>>>(Qb, Kb, Vb, Zb);

  // mid = 2 * (Zb x W_O^T)  -> d_out (f32) + midb (bf16)
  gemm_bf16_wmma<<<dim3(D_ / 64, M / 128), 128, 0, stream>>>(
      Zb, HDH_, Wob, HDH_, D_, HDH_, nullptr, 2.0f, 0, out, 0, midb);

  // h = relu(mid x W_in^T + b_in) -> hb (bf16)
  gemm_bf16_wmma<<<dim3(DM_ / 64, M / 128), 128, 0, stream>>>(
      midb, D_, Winb, D_, DM_, D_, b_in, 1.0f, 1, nullptr, 0, hb);

  // d_out += h x W_out^T + b_out
  gemm_bf16_wmma<<<dim3(D_ / 64, M / 128), 128, 0, stream>>>(
      hb, DM_, Woutb, DM_, D_, DM_, b_out, 1.0f, 0, out, 1, nullptr);
}
